// FP8Linear_67697274520372
// MI455X (gfx1250) — 8-bit matrix operand order repaired and checked by reading, compiled, not run on hardware
//
#include <hip/hip_runtime.h>
#include <hip/hip_bf16.h>
#include <stdint.h>

typedef __attribute__((ext_vector_type(16))) int   v16i;
typedef __attribute__((ext_vector_type(8)))  float v8f;
typedef __attribute__((ext_vector_type(4)))  int   v4i_t;

#define E4M3_MAX 448.0f
#define BM 128
#define BN 128
#define BK 128
#define LDS_PITCH 144
#define KDIM 4096
#define NDIM 4096
#define KT   (KDIM / BK)

__global__ void fp8lin_init_ws(unsigned int* amax) {
    if (threadIdx.x < 2) amax[threadIdx.x] = 0u;
}

__global__ __launch_bounds__(256)
void fp8lin_absmax(const float* __restrict__ in, size_t n, unsigned int* __restrict__ amax) {
    __shared__ float red[256];
    float m = 0.f;
    for (size_t i = (size_t)blockIdx.x * blockDim.x + threadIdx.x; i < n;
         i += (size_t)gridDim.x * blockDim.x)
        m = fmaxf(m, fabsf(in[i]));
    red[threadIdx.x] = m;
    __syncthreads();
    for (int s = 128; s > 0; s >>= 1) {
        if ((int)threadIdx.x < s)
            red[threadIdx.x] = fmaxf(red[threadIdx.x], red[threadIdx.x + s]);
        __syncthreads();
    }
    if (threadIdx.x == 0)
        atomicMax(amax, __float_as_uint(red[0]));
}

#if __has_builtin(__builtin_amdgcn_cvt_pk_fp8_f32)
#define HAVE_HW_FP8_CVT 1
#else
#define HAVE_HW_FP8_CVT 0
__device__ inline unsigned int f32_to_e4m3_byte(float f) {
    uint32_t u    = __float_as_uint(f);
    uint32_t sign = (u >> 24) & 0x80u;
    uint32_t absu = u & 0x7fffffffu;
    if (absu == 0u) return sign;
    int      exp  = (int)(absu >> 23) - 127;
    uint32_t mant = absu & 0x7fffffu;
    uint32_t q;
    if (exp >= -6) {
        q = ((uint32_t)(exp + 7) << 3) | (mant >> 20);
        uint32_t rem = mant & 0xfffffu;
        if (rem > 0x80000u || (rem == 0x80000u && (q & 1u))) q++;
        if (q > 0x7eu) q = 0x7eu;
    } else {
        float af = __uint_as_float(absu);
        int   m  = (int)rintf(af * 512.0f);
        q = (uint32_t)m;
    }
    return sign | q;
}
#endif

__global__ __launch_bounds__(256)
void fp8lin_quant(const float* __restrict__ in, uint32_t* __restrict__ out,
                  size_t n4, const unsigned int* __restrict__ amax) {
    const float ax    = fmaxf(__uint_as_float(*amax), 1e-12f);
    const float scale = E4M3_MAX / ax;
    for (size_t i = (size_t)blockIdx.x * blockDim.x + threadIdx.x; i < n4;
         i += (size_t)gridDim.x * blockDim.x) {
        float4 v = ((const float4*)in)[i];
        float a = fminf(fmaxf(v.x * scale, -E4M3_MAX), E4M3_MAX);
        float b = fminf(fmaxf(v.y * scale, -E4M3_MAX), E4M3_MAX);
        float c = fminf(fmaxf(v.z * scale, -E4M3_MAX), E4M3_MAX);
        float d = fminf(fmaxf(v.w * scale, -E4M3_MAX), E4M3_MAX);
#if HAVE_HW_FP8_CVT
        int p = 0;
        p = __builtin_amdgcn_cvt_pk_fp8_f32(a, b, p, false);
        p = __builtin_amdgcn_cvt_pk_fp8_f32(c, d, p, true);
        out[i] = (uint32_t)p;
#else
        out[i] = f32_to_e4m3_byte(a) | (f32_to_e4m3_byte(b) << 8) |
                 (f32_to_e4m3_byte(c) << 16) | (f32_to_e4m3_byte(d) << 24);
#endif
    }
}

#if __has_builtin(__builtin_amdgcn_global_load_async_to_lds_b128) && \
    __has_builtin(__builtin_amdgcn_s_wait_asynccnt)
#define HAVE_ASYNC_LDS 1
typedef __attribute__((address_space(1))) v4i_t g_v4i;
typedef __attribute__((address_space(3))) v4i_t l_v4i;
#else
#define HAVE_ASYNC_LDS 0
#endif

__device__ __forceinline__ void fp8_compute_tile(const uint8_t* __restrict__ AsB,
                                                 const uint8_t* __restrict__ BsB,
                                                 int lane, int wm, int wn,
                                                 v8f acc[4][2]) {
    const int bcol  = lane & 15;
    const int khalf = lane >> 4;
    v16i bf[2];
#pragma unroll
    for (int nj = 0; nj < 2; ++nj) {
        const uint8_t* bbase = BsB + (wn * 32 + nj * 16 + bcol) * LDS_PITCH + khalf * 16;
#pragma unroll
        for (int c = 0; c < 4; ++c) {
            int4 q = *(const int4*)(bbase + c * 32);
            bf[nj][4 * c + 0] = q.x; bf[nj][4 * c + 1] = q.y;
            bf[nj][4 * c + 2] = q.z; bf[nj][4 * c + 3] = q.w;
        }
    }
    const int arow = lane & 15;
    const int koff = (lane >> 4) * 16;
#pragma unroll
    for (int mi = 0; mi < 4; ++mi) {
        const uint8_t* abase = AsB + (wm * 64 + mi * 16 + arow) * LDS_PITCH + koff;
        v16i af;
#pragma unroll
        for (int j = 0; j < 4; ++j) {
            int4 q = *(const int4*)(abase + j * 32);
            af[4 * j + 0] = q.x; af[4 * j + 1] = q.y;
            af[4 * j + 2] = q.z; af[4 * j + 3] = q.w;
        }
#pragma unroll
        for (int nj = 0; nj < 2; ++nj) {
            acc[mi][nj] = __builtin_amdgcn_wmma_f32_16x16x128_fp8_fp8(
                af, bf[nj], (short)0, acc[mi][nj],
                 false,  false);
        }
    }
}

__global__ __launch_bounds__(256)
void fp8lin_gemm_wmma(const uint8_t* __restrict__ Aq,
                      const uint8_t* __restrict__ Bq,
                      __hip_bfloat16* __restrict__ out,
                      const unsigned int* __restrict__ amax) {
#if HAVE_ASYNC_LDS
    __shared__ uint8_t As[2][BM * LDS_PITCH];
    __shared__ uint8_t Bs[2][BN * LDS_PITCH];
#else
    __shared__ uint8_t As[1][BM * LDS_PITCH];
    __shared__ uint8_t Bs[1][BN * LDS_PITCH];
#endif

    const int tid  = threadIdx.x;
    const int lane = tid & 31;
    const int wave = tid >> 5;
    const int wm   = wave & 1;
    const int wn   = wave >> 1;

    const int m0 = blockIdx.y * BM;
    const int n0 = blockIdx.x * BN;

    const float mult = (fmaxf(__uint_as_float(amax[0]), 1e-12f) * (1.0f / E4M3_MAX)) *
                       (fmaxf(__uint_as_float(amax[1]), 1e-12f) * (1.0f / E4M3_MAX));

    v8f acc[4][2];
#pragma unroll
    for (int i = 0; i < 4; ++i)
#pragma unroll
        for (int j = 0; j < 2; ++j)
#pragma unroll
            for (int e = 0; e < 8; ++e) acc[i][j][e] = 0.0f;

    const int ldRow = tid >> 3;
    const int ldCol = (tid & 7) * 16;

#if HAVE_ASYNC_LDS
    #define ISSUE_STAGE(kt_, st_)                                                         \
        do {                                                                              \
            const int kk = (kt_) * BK;                                                    \
            _Pragma("unroll")                                                             \
            for (int p = 0; p < 4; ++p) {                                                 \
                const int r = p * 32 + ldRow;                                             \
                __builtin_amdgcn_global_load_async_to_lds_b128(                           \
                    (g_v4i*)(Aq + (size_t)(m0 + r) * KDIM + kk + ldCol),                  \
                    (l_v4i*)&As[st_][r * LDS_PITCH + ldCol], 0, 0);                       \
                __builtin_amdgcn_global_load_async_to_lds_b128(                           \
                    (g_v4i*)(Bq + (size_t)(n0 + r) * KDIM + kk + ldCol),                  \
                    (l_v4i*)&Bs[st_][r * LDS_PITCH + ldCol], 0, 0);                       \
            }                                                                             \
        } while (0)

    ISSUE_STAGE(0, 0);
    for (int kt = 0; kt < KT; ++kt) {
        const int st = kt & 1;
        if (kt + 1 < KT) {
            ISSUE_STAGE(kt + 1, (kt + 1) & 1);
            __builtin_amdgcn_s_wait_asynccnt(8);
        } else {
            __builtin_amdgcn_s_wait_asynccnt(0);
        }
        __syncthreads();
        fp8_compute_tile(&As[st][0], &Bs[st][0], lane, wm, wn, acc);
        __syncthreads();
    }
    #undef ISSUE_STAGE
#else
    for (int k0 = 0; k0 < KDIM; k0 += BK) {
#pragma unroll
        for (int p = 0; p < 4; ++p) {
            const int r = p * 32 + ldRow;
            uint4 va = *(const uint4*)(Aq + (size_t)(m0 + r) * KDIM + k0 + ldCol);
            *(uint4*)(&As[0][r * LDS_PITCH + ldCol]) = va;
            uint4 vb = *(const uint4*)(Bq + (size_t)(n0 + r) * KDIM + k0 + ldCol);
            *(uint4*)(&Bs[0][r * LDS_PITCH + ldCol]) = vb;
        }
        if (k0 + BK < KDIM) {
            __builtin_prefetch(Aq + (size_t)(m0 + ldRow) * KDIM + k0 + BK + ldCol, 0, 1);
            __builtin_prefetch(Bq + (size_t)(n0 + ldRow) * KDIM + k0 + BK + ldCol, 0, 1);
        }
        __syncthreads();
        fp8_compute_tile(&As[0][0], &Bs[0][0], lane, wm, wn, acc);
        __syncthreads();
    }
#endif

    const int ocol = lane & 15;
    const int orow = (lane >> 4) * 8;
#pragma unroll
    for (int mi = 0; mi < 4; ++mi)
#pragma unroll
        for (int nj = 0; nj < 2; ++nj) {
            const int gr0 = m0 + wm * 64 + mi * 16 + orow;
            const int gc  = n0 + wn * 32 + nj * 16 + ocol;
#pragma unroll
            for (int e = 0; e < 8; ++e)
                out[(size_t)(gr0 + e) * NDIM + gc] = __float2bfloat16(acc[mi][nj][e] * mult);
        }
}

extern "C" void kernel_launch(void* const* d_in, const int* in_sizes, int n_in,
                              void* d_out, int out_size, void* d_ws, size_t ws_size,
                              hipStream_t stream) {
    const float* x = (const float*)d_in[0];
    const float* w = (const float*)d_in[1];
    const size_t nx = (size_t)in_sizes[0];
    const size_t nw = (size_t)in_sizes[1];
    const int M = (int)(nx / KDIM);

    unsigned int* amax = (unsigned int*)d_ws;
    uint8_t* xq = (uint8_t*)d_ws + 256;
    uint8_t* wq = xq + nx;
    __hip_bfloat16* out = (__hip_bfloat16*)d_out;

    fp8lin_init_ws<<<1, 64, 0, stream>>>(amax);
    fp8lin_absmax<<<1024, 256, 0, stream>>>(x, nx, amax + 0);
    fp8lin_absmax<<<1024, 256, 0, stream>>>(w, nw, amax + 1);
    fp8lin_quant<<<2048, 256, 0, stream>>>(x, (uint32_t*)xq, nx / 4, amax + 0);
    fp8lin_quant<<<1024, 256, 0, stream>>>(w, (uint32_t*)wq, nw / 4, amax + 1);

    dim3 grid(NDIM / BN, M / BM);
    fp8lin_gemm_wmma<<<grid, 256, 0, stream>>>(xq, wq, out, amax);
}
